// Decoder_22574348108289
// MI455X (gfx1250) — compile-verified, matrix operand order checked by reading, not run on hardware
//
#include <hip/hip_runtime.h>

// ---------------------------------------------------------------------------
// GRU decoder for MI455X (gfx1250, wave32, WMMA bf16 16x16x32 w/ f32 accum)
//
// Phases:
//  1. one-time bf16 conversion of weights (+ Wout transpose) and emb table
//  2. tiny prep: ctx gather, style emb, h0 = [ctx,style]@Wp.T, gi1_const
//  3. parallel WMMA GEMM (2Mx2N register blocking): gi1_all = emb @ Wih1E.T
//  4. persistent cooperative kernel: 256 sequential GRU steps; per n-tile a
//     wave computes both M-tiles sharing one weight fragment
//  5. parallel WMMA GEMM (2Mx5N register blocking): logits = h1s @ WoutT.T
//  6. per-row log-softmax / argmax / target gather
// ---------------------------------------------------------------------------

typedef __attribute__((ext_vector_type(16))) __bf16 v16bf;
typedef __attribute__((ext_vector_type(8)))  float  v8f;
typedef unsigned short u16;

#define Bz 32
#define Tz 256
#define Hz 1024
#define Vz 10000
#define Ez 512
#define SEz 8
#define Cz 1024
#define G3H 3072          // 3*H
#define ROWS 8192         // T*B

// ---- workspace layout (bytes, all 256B aligned) ----------------------------
static constexpr size_t OFF_BAR  = 0;                                    // 256 B
static constexpr size_t OFF_A32  = 256;                                  // 32*1024*4
static constexpr size_t OFF_B32  = OFF_A32  + (size_t)Bz*Hz*4;
static constexpr size_t OFF_A16  = OFF_B32  + (size_t)Bz*Hz*4;
static constexpr size_t OFF_B16  = OFF_A16  + (size_t)Bz*Hz*2;
static constexpr size_t OFF_GH1  = OFF_B16  + (size_t)Bz*Hz*2;
static constexpr size_t OFF_GH2  = OFF_GH1  + (size_t)Bz*G3H*4;
static constexpr size_t OFF_GI2  = OFF_GH2  + (size_t)Bz*G3H*4;
static constexpr size_t OFF_GIC  = OFF_GI2  + (size_t)Bz*G3H*4;
static constexpr size_t OFF_CTX  = OFF_GIC  + (size_t)Bz*G3H*4;
static constexpr size_t OFF_SE   = OFF_CTX  + (size_t)Bz*Cz*4;
static constexpr size_t OFF_EMB  = OFF_SE   + (size_t)Bz*SEz*4;          // bf16 emb table
static constexpr size_t OFF_W1E  = OFF_EMB  + (size_t)Vz*Ez*2;           // Wih1[:, :E] bf16
static constexpr size_t OFF_WHH1 = OFF_W1E  + (size_t)G3H*Ez*2;
static constexpr size_t OFF_WIH2 = OFF_WHH1 + (size_t)G3H*Hz*2;
static constexpr size_t OFF_WHH2 = OFF_WIH2 + (size_t)G3H*Hz*2;
static constexpr size_t OFF_WOT  = OFF_WHH2 + (size_t)G3H*Hz*2;          // Wout^T [V,H] bf16
static constexpr size_t OFF_H1S  = OFF_WOT  + (size_t)Vz*Hz*2;           // h1s bf16 [8192,1024]
static constexpr size_t OFF_GI1  = OFF_H1S  + (size_t)ROWS*Hz*2;         // gi1_all f32 [8192,3072]

// ---- helpers ---------------------------------------------------------------
__device__ __forceinline__ u16 f2bf(float f) {
  unsigned int u = __float_as_uint(f);
  unsigned int r = (u + 0x7fffu + ((u >> 16) & 1u)) >> 16;
  return (u16)r;
}
__device__ __forceinline__ v16bf ldfrag(const u16* p) {
  return *reinterpret_cast<const v16bf*>(p);
}
__device__ __forceinline__ v8f wmma_bf16(v16bf a, v16bf b, v8f c) {
  return __builtin_amdgcn_wmma_f32_16x16x32_bf16(false, a, false, b, (short)0, c, false, false);
}
__device__ __forceinline__ float sigmoidf_(float x) { return 1.0f / (1.0f + __expf(-x)); }

// grid-wide sense-reversing barrier (bar[0]=count, bar[1]=generation)
__device__ __forceinline__ void grid_barrier(int* bar, int nblocks) {
  __threadfence();
  __syncthreads();
  if (threadIdx.x == 0) {
    int gen  = __hip_atomic_load(&bar[1], __ATOMIC_ACQUIRE, __HIP_MEMORY_SCOPE_AGENT);
    int prev = __hip_atomic_fetch_add(&bar[0], 1, __ATOMIC_ACQ_REL, __HIP_MEMORY_SCOPE_AGENT);
    if (prev == nblocks - 1) {
      __hip_atomic_store(&bar[0], 0, __ATOMIC_RELAXED, __HIP_MEMORY_SCOPE_AGENT);
      __hip_atomic_store(&bar[1], gen + 1, __ATOMIC_RELEASE, __HIP_MEMORY_SCOPE_AGENT);
    } else {
      while (__hip_atomic_load(&bar[1], __ATOMIC_ACQUIRE, __HIP_MEMORY_SCOPE_AGENT) == gen)
        __builtin_amdgcn_s_sleep(1);
    }
  }
  __syncthreads();
}

// Both 16-row M-tiles (M=32 total) for one 16-col N-tile, K=1024.
// A:[32,1024] bf16 row-major, W:[N,1024] bf16 row-major; shares the W fragment
// between the two WMMAs -> 3 fragment loads per 2 WMMAs.
__device__ __forceinline__ void wmma_unit_m32(const u16* A, const u16* W,
                                              float* out, int n0, int ldo) {
  const int lane = threadIdx.x & 31;
  const int ll = lane & 15;
  const int lhk = (lane >> 4) << 4;   // K offset 0 or 16
  v8f acc0 = {}, acc1 = {};
  const u16* ap = A + (size_t)ll * Hz + lhk;
  const u16* wp = W + (size_t)(n0 + ll) * Hz + lhk;
#pragma unroll 4
  for (int k0 = 0; k0 < Hz; k0 += 32) {
    v16bf bb = ldfrag(wp + k0);
    v16bf a0 = ldfrag(ap + k0);
    v16bf a1 = ldfrag(ap + (size_t)16 * Hz + k0);
    acc0 = wmma_bf16(a0, bb, acc0);
    acc1 = wmma_bf16(a1, bb, acc1);
  }
  const int lhm = (lane >> 4) << 3;  // 0 or 8
  float* o0 = out + (size_t)lhm * ldo + n0 + ll;
  float* o1 = out + (size_t)(16 + lhm) * ldo + n0 + ll;
#pragma unroll
  for (int v = 0; v < 8; ++v) {
    o0[(size_t)v * ldo] = acc0[v];
    o1[(size_t)v * ldo] = acc1[v];
  }
}

// ---- conversion kernels ----------------------------------------------------
__global__ void k_convert(const float* __restrict__ src, u16* __restrict__ dst, int n) {
  int i = blockIdx.x * blockDim.x + threadIdx.x;
  if (i < n) dst[i] = f2bf(src[i]);
}
// Wih1 [3072,1544] -> first 512 cols, bf16 [3072,512]
__global__ void k_convert_w1e(const float* __restrict__ Wih1, u16* __restrict__ dst) {
  int i = blockIdx.x * blockDim.x + threadIdx.x;   // 3072*512
  int j = i >> 9, k = i & 511;
  dst[i] = f2bf(Wih1[(size_t)j * 1544 + k]);
}
// Wout [1024,10000] -> transposed bf16 [10000,1024]
__global__ void k_convert_woutT(const float* __restrict__ Wout, u16* __restrict__ dst) {
  int i = blockIdx.x * blockDim.x + threadIdx.x;   // 10000*1024
  int v = i >> 10, h = i & 1023;
  dst[i] = f2bf(Wout[(size_t)h * Vz + v]);
}

// ---- prep: ctx gather + style emb ------------------------------------------
__global__ void k_prep_ctx(const float* __restrict__ enc_output, const int* __restrict__ enc_len,
                           const int* __restrict__ styles, const float* __restrict__ style_table,
                           float* __restrict__ ctx, float* __restrict__ se) {
  for (int i = threadIdx.x; i < Bz * Cz; i += blockDim.x) {
    int b = i >> 10, k = i & 1023;
    int last = enc_len[b] - 1;
    last = last < 0 ? 0 : (last > Tz - 1 ? Tz - 1 : last);
    ctx[i] = enc_output[((size_t)b * 256 + last) * Cz + k];
  }
  for (int i = threadIdx.x; i < Bz * SEz; i += blockDim.x) {
    int b = i >> 3, k = i & 7;
    se[i] = style_table[styles[b] * SEz + k];
  }
}

// ---- prep: h0 = [ctx,style]@Wp.T + bp ; gi1_const = [style,ctx]@Wih1[:,E:].T + bih1
__global__ void k_prep_proj(const float* __restrict__ Wp, const float* __restrict__ bp,
                            const float* __restrict__ Wih1, const float* __restrict__ bih1,
                            const float* __restrict__ ctx, const float* __restrict__ se,
                            float* __restrict__ h0, float* __restrict__ gic) {
  int idx = blockIdx.x * blockDim.x + threadIdx.x;     // 512 blocks * 256 = 131072
  if (idx < Bz * Hz) {
    int b = idx >> 10, h = idx & 1023;
    const float* wr = Wp + (size_t)h * (Cz + SEz);     // cols: [ctx(1024), style(8)]
    const float* cb = ctx + b * Cz;
    const float* sb = se + b * SEz;
    float acc = bp[h];
    for (int k = 0; k < Cz; ++k) acc += cb[k] * wr[k];
    for (int k = 0; k < SEz; ++k) acc += sb[k] * wr[Cz + k];
    h0[idx] = acc;
  } else {
    int i = idx - Bz * Hz;                             // 0 .. 98303
    int b = i / G3H, j = i % G3H;
    const float* wr = Wih1 + (size_t)j * 1544;         // x = [emb(512), style(8), ctx(1024)]
    const float* cb = ctx + b * Cz;
    const float* sb = se + b * SEz;
    float acc = bih1[j];
    for (int k = 0; k < SEz; ++k) acc += sb[k] * wr[Ez + k];
    for (int k = 0; k < Cz; ++k) acc += cb[k] * wr[Ez + SEz + k];
    gic[(size_t)b * G3H + j] = acc;
  }
}

// ---- gi1_all = gather(emb, dec_in) @ Wih1E.T  (K=512, 2Mx2N blocking) ------
__global__ void k_gi1_gemm(const int* __restrict__ dec_in, const u16* __restrict__ emb16,
                           const u16* __restrict__ w1e, float* __restrict__ gi1_all) {
  int wave = (blockIdx.x * blockDim.x + threadIdx.x) >> 5;   // 0..24575
  const int lane = threadIdx.x & 31;
  const int ll = lane & 15;
  const int lhk = (lane >> 4) << 4;
  int mg = wave & 255;              // 256 groups of 32 rows (r = t*32+b order)
  int ng = wave >> 8;               // 96 groups of 32 cols
  int r0 = mg << 5;                 // first row of group
  int n0 = ng << 5;                 // first col of group
  int ra = r0 + ll, rb = r0 + 16 + ll;
  int tok0 = dec_in[(ra & 31) * Tz + (ra >> 5)];
  int tok1 = dec_in[(rb & 31) * Tz + (rb >> 5)];
  const u16* ap0 = emb16 + (size_t)tok0 * Ez + lhk;
  const u16* ap1 = emb16 + (size_t)tok1 * Ez + lhk;
  const u16* wp0 = w1e + (size_t)(n0 + ll) * Ez + lhk;
  const u16* wp1 = w1e + (size_t)(n0 + 16 + ll) * Ez + lhk;
  v8f acc00 = {}, acc01 = {}, acc10 = {}, acc11 = {};
#pragma unroll 4
  for (int k0 = 0; k0 < Ez; k0 += 32) {
    v16bf a0 = ldfrag(ap0 + k0);
    v16bf a1 = ldfrag(ap1 + k0);
    v16bf b0 = ldfrag(wp0 + k0);
    v16bf b1 = ldfrag(wp1 + k0);
    acc00 = wmma_bf16(a0, b0, acc00);
    acc01 = wmma_bf16(a0, b1, acc01);
    acc10 = wmma_bf16(a1, b0, acc10);
    acc11 = wmma_bf16(a1, b1, acc11);
  }
  const int lhm = (lane >> 4) << 3;
  v8f accs[2][2] = {{acc00, acc01}, {acc10, acc11}};
#pragma unroll
  for (int i = 0; i < 2; ++i) {
#pragma unroll
    for (int j = 0; j < 2; ++j) {
      float* orow = gi1_all + (size_t)(r0 + i * 16 + lhm) * G3H + n0 + j * 16 + ll;
#pragma unroll
      for (int v = 0; v < 8; ++v) orow[(size_t)v * G3H] = accs[i][j][v];
    }
  }
}

// ---- persistent recurrence kernel ------------------------------------------
#define REC_BLOCKS 48
#define REC_THREADS 256
#define REC_WAVES (REC_BLOCKS * REC_THREADS / 32)     // 384
#define REC_TOT (REC_BLOCKS * REC_THREADS)            // 12288

__global__ void k_recurrence(const float* __restrict__ gi1_all, const float* __restrict__ gic,
                             const u16* __restrict__ whh1, const u16* __restrict__ wih2,
                             const u16* __restrict__ whh2,
                             const float* __restrict__ bhh1, const float* __restrict__ bih2,
                             const float* __restrict__ bhh2,
                             float* __restrict__ a32, u16* __restrict__ a16,
                             float* __restrict__ b32, u16* __restrict__ b16,
                             float* __restrict__ gh1, float* __restrict__ gh2,
                             float* __restrict__ gi2, u16* __restrict__ h1s,
                             int* bar) {
  const int gtid = blockIdx.x * blockDim.x + threadIdx.x;
  const int wave = gtid >> 5;

  // init: a16 <- bf16(h0), b16 <- bf16(0)
  for (int i = gtid; i < Bz * Hz; i += REC_TOT) {
    a16[i] = f2bf(a32[i]);
    b16[i] = f2bf(b32[i]);
  }
  grid_barrier(bar, REC_BLOCKS);

  for (int t = 0; t < Tz; ++t) {
    // ---- phase A: gh1 = a@Whh1.T ; gh2 = b@Whh2.T  (2 gemms x 192 n-tiles) --
    // one unit per wave: both M-tiles of one n-tile, shared weight fragment
    {
      int g = wave >= 192;                     // 0 -> gh1, 1 -> gh2
      int n0 = (g ? wave - 192 : wave) << 4;
      wmma_unit_m32(g ? b16 : a16, g ? whh2 : whh1, g ? gh2 : gh1, n0, G3H);
    }
    grid_barrier(bar, REC_BLOCKS);

    // ---- E1: a = GRU1(gi1_all[t] + gic, gh1 + bhh1, a) ----
    for (int i = gtid; i < Bz * Hz; i += REC_TOT) {
      int b = i >> 10, j = i & 1023;
      const float* gA = gi1_all + ((size_t)(t * Bz + b)) * G3H;
      const float* gC = gic + (size_t)b * G3H;
      const float* gH = gh1 + (size_t)b * G3H;
      float ir  = gA[j]        + gC[j];
      float iz  = gA[Hz + j]   + gC[Hz + j];
      float inn = gA[2*Hz + j] + gC[2*Hz + j];
      float hr  = gH[j]        + bhh1[j];
      float hz  = gH[Hz + j]   + bhh1[Hz + j];
      float hn  = gH[2*Hz + j] + bhh1[2*Hz + j];
      float r = sigmoidf_(ir + hr);
      float z = sigmoidf_(iz + hz);
      float n = tanhf(inn + r * hn);
      float an = (1.0f - z) * n + z * a32[i];
      a32[i] = an;
      a16[i] = f2bf(an);
    }
    grid_barrier(bar, REC_BLOCKS);

    // ---- phase B: gi2 = a_new @ Wih2.T  (192 n-tile units) ----
    if (wave < 192) {
      wmma_unit_m32(a16, wih2, gi2, wave << 4, G3H);
    }
    grid_barrier(bar, REC_BLOCKS);

    // ---- E2: b = GRU2(gi2 + bih2, gh2 + bhh2, b); h1s[t] = bf16(b) ----
    for (int i = gtid; i < Bz * Hz; i += REC_TOT) {
      int b = i >> 10, j = i & 1023;
      const float* gI = gi2 + (size_t)b * G3H;
      const float* gH = gh2 + (size_t)b * G3H;
      float ir  = gI[j]        + bih2[j];
      float iz  = gI[Hz + j]   + bih2[Hz + j];
      float inn = gI[2*Hz + j] + bih2[2*Hz + j];
      float hr  = gH[j]        + bhh2[j];
      float hz  = gH[Hz + j]   + bhh2[Hz + j];
      float hn  = gH[2*Hz + j] + bhh2[2*Hz + j];
      float r = sigmoidf_(ir + hr);
      float z = sigmoidf_(iz + hz);
      float n = tanhf(inn + r * hn);
      float bn = (1.0f - z) * n + z * b32[i];
      b32[i] = bn;
      u16 bh = f2bf(bn);
      b16[i] = bh;
      h1s[((size_t)(t * Bz + b)) * Hz + j] = bh;
    }
    grid_barrier(bar, REC_BLOCKS);
  }
}

// ---- logits = h1s @ WoutT.T + bout (M=8192, N=10000, K=1024) ---------------
// 2Mx5N register blocking: 7 fragment loads per 10 WMMAs (~23 FLOP/B from L2)
__global__ void k_logits_gemm(const u16* __restrict__ h1s, const u16* __restrict__ woutT,
                              const float* __restrict__ bout, float* __restrict__ logits) {
  int wave = (blockIdx.x * blockDim.x + threadIdx.x) >> 5;  // 0..31999
  const int lane = threadIdx.x & 31;
  const int ll = lane & 15;
  const int lhk = (lane >> 4) << 4;
  int ng = wave / 256;           // 0..124 -> 5 n-tiles (80 cols)
  int mg = wave % 256;           // 0..255 -> 2 m-tiles (32 rows)
  int n0 = ng * 80;
  int r0 = mg << 5;
  const u16* ap = h1s + (size_t)(r0 + ll) * Hz + lhk;
  v8f acc[2][5];
#pragma unroll
  for (int i = 0; i < 2; ++i)
#pragma unroll
    for (int j = 0; j < 5; ++j) acc[i][j] = (v8f){};
#pragma unroll 2
  for (int k0 = 0; k0 < Hz; k0 += 32) {
    v16bf a0 = ldfrag(ap + k0);
    v16bf a1 = ldfrag(ap + (size_t)16 * Hz + k0);
#pragma unroll
    for (int j = 0; j < 5; ++j) {
      v16bf bb = ldfrag(woutT + (size_t)(n0 + j * 16 + ll) * Hz + lhk + k0);
      acc[0][j] = wmma_bf16(a0, bb, acc[0][j]);
      acc[1][j] = wmma_bf16(a1, bb, acc[1][j]);
    }
  }
  const int lhm = (lane >> 4) << 3;
#pragma unroll
  for (int j = 0; j < 5; ++j) {
    int n = n0 + j * 16 + ll;
    float bias = bout[n];
#pragma unroll
    for (int i = 0; i < 2; ++i) {
#pragma unroll
      for (int v = 0; v < 8; ++v) {
        int r = r0 + i * 16 + lhm + v;       // row in (t*32+b) order
        int t = r >> 5, b = r & 31;
        logits[((size_t)(b * Tz + t)) * Vz + n] = acc[i][j][v] + bias;
      }
    }
  }
}

// ---- log-softmax / argmax / target gather, one block per (b,t) row ---------
__global__ void k_softmax(const float* __restrict__ logits, const int* __restrict__ dec_out,
                          float* __restrict__ out_lp, float* __restrict__ out_pred) {
  __shared__ float sv[256];
  __shared__ int   si[256];
  __shared__ float ss[256];
  int row = blockIdx.x;                         // = b*256 + t
  const float* lg = logits + (size_t)row * Vz;
  float m = -3.4e38f; int am = 0;
  for (int v = threadIdx.x; v < Vz; v += 256) {
    float x = lg[v];
    if (x > m) { m = x; am = v; }
  }
  sv[threadIdx.x] = m; si[threadIdx.x] = am;
  __syncthreads();
  for (int s = 128; s > 0; s >>= 1) {
    if (threadIdx.x < s) {
      float ov = sv[threadIdx.x + s]; int oi = si[threadIdx.x + s];
      if (ov > sv[threadIdx.x] || (ov == sv[threadIdx.x] && oi < si[threadIdx.x])) {
        sv[threadIdx.x] = ov; si[threadIdx.x] = oi;
      }
    }
    __syncthreads();
  }
  float M = sv[0]; int AM = si[0];
  float sum = 0.0f;
  for (int v = threadIdx.x; v < Vz; v += 256) sum += __expf(lg[v] - M);
  ss[threadIdx.x] = sum;
  __syncthreads();
  for (int s = 128; s > 0; s >>= 1) {
    if (threadIdx.x < s) ss[threadIdx.x] += ss[threadIdx.x + s];
    __syncthreads();
  }
  if (threadIdx.x == 0) {
    float logZ = M + __logf(ss[0]);
    int tgt = dec_out[row];
    out_lp[row]   = lg[tgt] - logZ;
    out_pred[row] = (float)AM;
  }
}

// ---------------------------------------------------------------------------
extern "C" void kernel_launch(void* const* d_in, const int* in_sizes, int n_in,
                              void* d_out, int out_size, void* d_ws, size_t ws_size,
                              hipStream_t stream) {
  (void)in_sizes; (void)n_in; (void)out_size; (void)ws_size;
  const float* enc_output  = (const float*)d_in[0];
  const int*   enc_len     = (const int*)d_in[1];
  const int*   styles      = (const int*)d_in[2];
  const int*   dec_in      = (const int*)d_in[3];
  const int*   dec_out     = (const int*)d_in[4];
  const float* emb_table   = (const float*)d_in[5];
  const float* style_table = (const float*)d_in[6];
  const float* Wp   = (const float*)d_in[7];
  const float* bp   = (const float*)d_in[8];
  const float* Wih1 = (const float*)d_in[9];
  const float* Whh1 = (const float*)d_in[10];
  const float* bih1 = (const float*)d_in[11];
  const float* bhh1 = (const float*)d_in[12];
  const float* Wih2 = (const float*)d_in[13];
  const float* Whh2 = (const float*)d_in[14];
  const float* bih2 = (const float*)d_in[15];
  const float* bhh2 = (const float*)d_in[16];
  const float* Wout = (const float*)d_in[17];
  const float* bout = (const float*)d_in[18];

  char* w = (char*)d_ws;
  float* a32 = (float*)(w + OFF_A32);
  float* b32 = (float*)(w + OFF_B32);
  u16*   a16 = (u16*)(w + OFF_A16);
  u16*   b16 = (u16*)(w + OFF_B16);
  float* gh1 = (float*)(w + OFF_GH1);
  float* gh2 = (float*)(w + OFF_GH2);
  float* gi2 = (float*)(w + OFF_GI2);
  float* gic = (float*)(w + OFF_GIC);
  float* ctx = (float*)(w + OFF_CTX);
  float* sem = (float*)(w + OFF_SE);
  u16*   emb16  = (u16*)(w + OFF_EMB);
  u16*   w1e16  = (u16*)(w + OFF_W1E);
  u16*   whh116 = (u16*)(w + OFF_WHH1);
  u16*   wih216 = (u16*)(w + OFF_WIH2);
  u16*   whh216 = (u16*)(w + OFF_WHH2);
  u16*   wout16 = (u16*)(w + OFF_WOT);
  u16*   h1s16  = (u16*)(w + OFF_H1S);
  float* gi1all = (float*)(w + OFF_GI1);
  int*   bar    = (int*)(w + OFF_BAR);

  float* logits = (float*)d_out;
  float* out_lp = logits + (size_t)Bz * Tz * Vz;       // 81,920,000
  float* out_pr = out_lp + (size_t)Bz * Tz;            // +8192

  // deterministic state init
  hipMemsetAsync(w + OFF_BAR, 0, 256, stream);
  hipMemsetAsync(w + OFF_B32, 0, (size_t)Bz * Hz * 4, stream);

  // one-time bf16 conversions
  k_convert<<<dim3((Vz * Ez) / 256), dim3(256), 0, stream>>>(emb_table, emb16, Vz * Ez);
  k_convert_w1e<<<dim3((G3H * Ez) / 256), dim3(256), 0, stream>>>(Wih1, w1e16);
  k_convert<<<dim3((G3H * Hz) / 256), dim3(256), 0, stream>>>(Whh1, whh116, G3H * Hz);
  k_convert<<<dim3((G3H * Hz) / 256), dim3(256), 0, stream>>>(Wih2, wih216, G3H * Hz);
  k_convert<<<dim3((G3H * Hz) / 256), dim3(256), 0, stream>>>(Whh2, whh216, G3H * Hz);
  k_convert_woutT<<<dim3((Vz * Hz) / 256), dim3(256), 0, stream>>>(Wout, wout16);

  // prep
  k_prep_ctx<<<dim3(1), dim3(256), 0, stream>>>(enc_output, enc_len, styles, style_table, ctx, sem);
  k_prep_proj<<<dim3(512), dim3(256), 0, stream>>>(Wp, bp, Wih1, bih1, ctx, sem, a32, gic);

  // big parallel GEMM for the time-varying part of cell-1 input (24576 waves)
  k_gi1_gemm<<<dim3(3072), dim3(256), 0, stream>>>(dec_in, emb16, w1e16, gi1all);

  // sequential recurrence (persistent cooperative kernel)
  k_recurrence<<<dim3(REC_BLOCKS), dim3(REC_THREADS), 0, stream>>>(
      gi1all, gic, whh116, wih216, whh216, bhh1, bih2, bhh2,
      a32, a16, b32, b16, gh1, gh2, gi2, h1s16, bar);

  // output projection (32000 waves) + softmax stats
  k_logits_gemm<<<dim3(4000), dim3(256), 0, stream>>>(h1s16, wout16, bout, logits);
  k_softmax<<<dim3(8192), dim3(256), 0, stream>>>(logits, dec_out, out_lp, out_pr);
}
